// GNNV2Model_29609504539151
// MI455X (gfx1250) — compile-verified, matrix operand order checked by reading, not run on hardware
//
#include <hip/hip_runtime.h>
#include <hip/hip_bf16.h>

// ---------------------------------------------------------------------------
// GINE GNN forward for MI455X (gfx1250, wave32, WMMA).
//   All dense layers -> f16 WMMA (v_wmma_f32_16x16x32_f16, f32 accumulate).
//   Edge features cached as f16 (halves the dominant HBM stream).
//   Scatter/segment-sum via hardware global_atomic_add_f32.
//   GELU is branch-free (v_exp_f32 + v_rcp_f32), no libm tanhf divergence.
// ---------------------------------------------------------------------------

typedef __attribute__((ext_vector_type(16))) _Float16 v16h;
typedef __attribute__((ext_vector_type(8)))  _Float16 v8h;
typedef __attribute__((ext_vector_type(4)))  _Float16 v4h;
typedef __attribute__((ext_vector_type(8)))  float    v8f;

#define ACT_NONE 0
#define ACT_RELU 1
#define ACT_GELU 2

static constexpr int NNODE = 50000;
static constexpr int NEDGE = 800000;
static constexpr int DFEAT = 128;
static constexpr int NGRAPH = 1024;
static constexpr int HEADH = 512;
static constexpr int NOUT = 3;

// jax.nn.gelu default (approximate=True), branch-free:
// tanh(z) = copysign((1-t)/(1+t), z), t = exp(-2|z|)  -> v_exp_f32 + v_rcp_f32
__device__ __forceinline__ float gelu_tanh(float x) {
    float x3 = x * x * x;
    float z = 0.7978845608028654f * (x + 0.044715f * x3);
    float t = __expf(-2.0f * fabsf(z));
    float th = __builtin_copysignf(__fdividef(1.0f - t, 1.0f + t), z);
    return 0.5f * x * (1.0f + th);
}

template <int ACT>
__device__ __forceinline__ float apply_act(float x) {
    if constexpr (ACT == ACT_RELU) return fmaxf(x, 0.0f);
    else if constexpr (ACT == ACT_GELU) return gelu_tanh(x);
    else return x;
}

// ---------------------------------------------------------------------------
// Tiled WMMA GEMM:  C[M, N] = act(A[M, KD] @ W[KD, N] + bias)
//   Block: 256 threads = 8 wave32.  Tile: 128 rows x 128 cols, BK = 32.
//   Wave w owns rows [w*16, w*16+16) of the tile -> 8 v8f accumulators.
//   A, W converted f32 -> f16 into LDS; stride 40 halves (80 B) keeps the
//   16-byte fragment reads bank-conflict free and 16 B aligned.
// ---------------------------------------------------------------------------
static constexpr int BM = 128, BN = 128, BK = 32;
static constexpr int ASTR = 40;  // halves per LDS row

template <int KD, int ACT, bool OUTF16>
__global__ __launch_bounds__(256) void gemm_bias_act(
    const float* __restrict__ A,      // [M, KD]
    const float* __restrict__ W,      // [KD, ldw]
    int ldw,                          // = total N
    const float* __restrict__ bias,   // [ldw]
    float* __restrict__ Cf,           // f32 out (or null)
    _Float16* __restrict__ Ch,        // f16 out (or null)
    int M) {
    __shared__ _Float16 Alds[BM * ASTR];
    __shared__ _Float16 Blds[BN * ASTR];

    const int tid = threadIdx.x;
    const int lane = tid & 31;
    const int wave = tid >> 5;
    const int blockRow = blockIdx.x * BM;
    const int colBase = blockIdx.y * BN;

    v8f acc[8] = {};

    constexpr int KT = (KD + BK - 1) / BK;
#pragma unroll
    for (int kt = 0; kt < KT; ++kt) {
        const int kbase = kt * BK;

        // ---- stage A tile (128 x 32 f32 -> f16), 4 float4 per thread ----
#pragma unroll
        for (int i = 0; i < 4; ++i) {
            int j = tid + i * 256;          // 0..1023 float4 slots
            int row = j >> 3;
            int c4 = (j & 7) * 4;
            int grow = blockRow + row;
            float4 v = make_float4(0.f, 0.f, 0.f, 0.f);
            if (grow < M && (kbase + c4) < KD)  // KD always multiple of 4
                v = *(const float4*)(A + (size_t)grow * KD + kbase + c4);
            _Float16* d = &Alds[row * ASTR + c4];
            d[0] = (_Float16)v.x; d[1] = (_Float16)v.y;
            d[2] = (_Float16)v.z; d[3] = (_Float16)v.w;
        }
        // ---- stage B tile transposed: Blds[col][k], 16 elems per thread ----
#pragma unroll
        for (int i = 0; i < 16; ++i) {
            int j = tid + i * 256;          // 0..4095
            int k = j >> 7;
            int col = j & 127;
            float w = 0.f;
            if (kbase + k < KD)
                w = W[(size_t)(kbase + k) * ldw + colBase + col];
            Blds[col * ASTR + k] = (_Float16)w;
        }
        // prefetch next A K-slab while we compute (global_prefetch_b8)
        if (kt + 1 < KT) {
            int prow = blockRow + (tid & 127);
            if (prow < M)
                __builtin_prefetch(A + (size_t)prow * KD + kbase + BK, 0, 0);
        }
        __syncthreads();

        // ---- A fragment: row = lane&15 of this wave's strip, K half by lane>>4
        const _Float16* ap =
            &Alds[(wave * 16 + (lane & 15)) * ASTR + (lane >> 4) * 16];
        v8h alo = *(const v8h*)ap;
        v8h ahi = *(const v8h*)(ap + 8);
        v16h afrag = __builtin_shufflevector(alo, ahi, 0, 1, 2, 3, 4, 5, 6, 7,
                                             8, 9, 10, 11, 12, 13, 14, 15);
#pragma unroll
        for (int c = 0; c < 8; ++c) {
            const _Float16* bp =
                &Blds[(c * 16 + (lane & 15)) * ASTR + (lane >> 4) * 16];
            v8h blo = *(const v8h*)bp;
            v8h bhi = *(const v8h*)(bp + 8);
            v16h bfrag = __builtin_shufflevector(blo, bhi, 0, 1, 2, 3, 4, 5, 6,
                                                 7, 8, 9, 10, 11, 12, 13, 14, 15);
            acc[c] = __builtin_amdgcn_wmma_f32_16x16x32_f16(
                false, afrag, false, bfrag, (short)0, acc[c], false, false);
        }
        __syncthreads();
    }

    // ---- epilogue: C layout -> VGPR r holds M=r (lanes 0-15) / M=r+8 (16-31)
    const int rbase = blockRow + wave * 16 + ((lane >> 4) ? 8 : 0);
    const int n = lane & 15;
#pragma unroll
    for (int c = 0; c < 8; ++c) {
        int col = colBase + c * 16 + n;
        float bv = bias[col];
#pragma unroll
        for (int r = 0; r < 8; ++r) {
            int row = rbase + r;
            if (row < M) {
                float x = apply_act<ACT>(acc[c][r] + bv);
                if constexpr (OUTF16)
                    Ch[(size_t)row * ldw + col] = (_Float16)x;
                else
                    Cf[(size_t)row * ldw + col] = x;
            }
        }
    }
}

// ---------------------------------------------------------------------------
// agg[i] = (1 + eps) * h[i]          (float4 granularity)
// ---------------------------------------------------------------------------
__global__ void scale_copy(const float* __restrict__ h,
                           const float* __restrict__ eps,
                           float* __restrict__ agg, int n4) {
    int i = blockIdx.x * blockDim.x + threadIdx.x;
    if (i >= n4) return;
    float s = 1.0f + eps[0];
    float4 v = ((const float4*)h)[i];
    v.x *= s; v.y *= s; v.z *= s; v.w *= s;
    ((float4*)agg)[i] = v;
}

// ---------------------------------------------------------------------------
// One wave per edge: m = relu(h[src] + e) ; agg[dst] += m  (atomic f32 add)
// lane handles float4 channel chunk `lane` (32 lanes * 4 = 128 channels).
// h table (25.6 MB) is L2-resident; e streams from HBM as f16.
// ---------------------------------------------------------------------------
__global__ __launch_bounds__(256) void gine_msg(
    const float* __restrict__ h, const _Float16* __restrict__ e,
    const int* __restrict__ src, const int* __restrict__ dst,
    float* __restrict__ agg, int E) {
    int gt = blockIdx.x * 256 + threadIdx.x;
    int edge = gt >> 5;
    if (edge >= E) return;
    int lane = gt & 31;
    int s = src[edge];
    int d = dst[edge];
    float4 hv = *(const float4*)(h + (size_t)s * DFEAT + lane * 4);
    v4h ev = *(const v4h*)(e + (size_t)edge * DFEAT + lane * 4);
    float m0 = fmaxf(hv.x + (float)ev[0], 0.0f);
    float m1 = fmaxf(hv.y + (float)ev[1], 0.0f);
    float m2 = fmaxf(hv.z + (float)ev[2], 0.0f);
    float m3 = fmaxf(hv.w + (float)ev[3], 0.0f);
    float* ap = agg + (size_t)d * DFEAT + lane * 4;
    unsafeAtomicAdd(ap + 0, m0);
    unsafeAtomicAdd(ap + 1, m1);
    unsafeAtomicAdd(ap + 2, m2);
    unsafeAtomicAdd(ap + 3, m3);
}

// ---------------------------------------------------------------------------
// global_add_pool: pooled[batch[node]] += hf[node]   (one wave per node)
// ---------------------------------------------------------------------------
__global__ __launch_bounds__(256) void pool_scatter(
    const float* __restrict__ hf, const int* __restrict__ batch,
    float* __restrict__ pooled, int N) {
    int gt = blockIdx.x * 256 + threadIdx.x;
    int node = gt >> 5;
    if (node >= N) return;
    int lane = gt & 31;
    int g = batch[node];
    float4 v = *(const float4*)(hf + (size_t)node * DFEAT + lane * 4);
    float* pp = pooled + (size_t)g * DFEAT + lane * 4;
    unsafeAtomicAdd(pp + 0, v.x);
    unsafeAtomicAdd(pp + 1, v.y);
    unsafeAtomicAdd(pp + 2, v.z);
    unsafeAtomicAdd(pp + 3, v.w);
}

__global__ void zero_f32(float* p, int n) {
    int i = blockIdx.x * blockDim.x + threadIdx.x;
    if (i < n) p[i] = 0.0f;
}

// ---------------------------------------------------------------------------
// Final tiny projection: logits[g, o] = relu_mid[g, :] @ w[:, o] + b[o]
// (1024 x 512 x 3 = 3 MFLOP, negligible; plain VALU)
// ---------------------------------------------------------------------------
__global__ void head_out(const float* __restrict__ mid,
                         const float* __restrict__ w,
                         const float* __restrict__ b,
                         float* __restrict__ out) {
    int gt = blockIdx.x * blockDim.x + threadIdx.x;
    if (gt >= NGRAPH * NOUT) return;
    int g = gt / NOUT, o = gt % NOUT;
    float s = b[o];
    const float* mrow = mid + (size_t)g * HEADH;
#pragma unroll 4
    for (int k = 0; k < HEADH; ++k) s += mrow[k] * w[k * NOUT + o];
    out[gt] = s;
}

// ---------------------------------------------------------------------------
extern "C" void kernel_launch(void* const* d_in, const int* in_sizes, int n_in,
                              void* d_out, int out_size, void* d_ws,
                              size_t ws_size, hipStream_t stream) {
    (void)in_sizes; (void)n_in; (void)out_size; (void)ws_size;
    // --- inputs (setup_inputs() insertion order, params depth-first) ---
    const float* x       = (const float*)d_in[0];   // [N, 64]
    const int*   eidx    = (const int*)  d_in[1];   // [2, E]
    const float* eattr   = (const float*)d_in[2];   // [E, 16]
    const int*   batch   = (const int*)  d_in[3];   // [N]
    const float* node_w0 = (const float*)d_in[4];
    const float* node_b0 = (const float*)d_in[5];
    const float* node_w1 = (const float*)d_in[6];
    const float* node_b1 = (const float*)d_in[7];
    const float* edge_w0 = (const float*)d_in[8];
    const float* edge_b0 = (const float*)d_in[9];
    const float* edge_w1 = (const float*)d_in[10];
    const float* edge_b1 = (const float*)d_in[11];
    // convs[i]: eps, w1, b1, w2, b2 at 12 + 5*i
    const float* lin_w   = (const float*)d_in[32];
    const float* lin_b   = (const float*)d_in[33];
    const float* head_w0 = (const float*)d_in[34];
    const float* head_b0 = (const float*)d_in[35];
    const float* head_w1 = (const float*)d_in[36];
    const float* head_b1 = (const float*)d_in[37];

    const int N = NNODE, E = NEDGE;
    const int* src = eidx;
    const int* dst = eidx + E;

    // --- workspace carve (256 B aligned) ---
    char* wsp = (char*)d_ws;
    auto carve = [&](size_t bytes) {
        void* p = (void*)wsp;
        wsp += (bytes + 255) & ~(size_t)255;
        return p;
    };
    _Float16* e16  = (_Float16*)carve((size_t)E * DFEAT * sizeof(_Float16));
    float*    emid = (float*)   carve((size_t)E * DFEAT * sizeof(float));
    float*    h    = (float*)   carve((size_t)N * DFEAT * sizeof(float));
    float*    t0   = (float*)   carve((size_t)N * DFEAT * sizeof(float));
    float*    agg  = (float*)   carve((size_t)N * DFEAT * sizeof(float));
    float*    pool = (float*)   carve((size_t)NGRAPH * DFEAT * sizeof(float));
    float*    hmid = (float*)   carve((size_t)NGRAPH * HEADH * sizeof(float));

    const dim3 blk(256);
    const dim3 gE((E + BM - 1) / BM, 1);           // 6250 row-tiles
    const dim3 gN((N + BM - 1) / BM, 1);           // 391 row-tiles
    const int warpBlocksE = (E * 32 + 255) / 256;  // one wave per edge
    const int warpBlocksN = (N * 32 + 255) / 256;  // one wave per node

    // --- edge MLP: gelu(gelu(ea@w0+b0)@w1+b1) -> e16 (f16, streamed 4x) ---
    gemm_bias_act<16, ACT_GELU, false><<<gE, blk, 0, stream>>>(
        eattr, edge_w0, DFEAT, edge_b0, emid, nullptr, E);
    gemm_bias_act<128, ACT_GELU, true><<<gE, blk, 0, stream>>>(
        emid, edge_w1, DFEAT, edge_b1, nullptr, e16, E);

    // --- node MLP: gelu(gelu(x@w0+b0)@w1+b1) -> h ---
    gemm_bias_act<64, ACT_GELU, false><<<gN, blk, 0, stream>>>(
        x, node_w0, DFEAT, node_b0, t0, nullptr, N);
    gemm_bias_act<128, ACT_GELU, false><<<gN, blk, 0, stream>>>(
        t0, node_w1, DFEAT, node_b1, h, nullptr, N);

    // --- 4 GINEConv layers ---
    for (int i = 0; i < 4; ++i) {
        const float* eps = (const float*)d_in[12 + 5 * i];
        const float* w1  = (const float*)d_in[13 + 5 * i];
        const float* b1  = (const float*)d_in[14 + 5 * i];
        const float* w2  = (const float*)d_in[15 + 5 * i];
        const float* b2  = (const float*)d_in[16 + 5 * i];

        int n4 = N * DFEAT / 4;
        scale_copy<<<(n4 + 255) / 256, blk, 0, stream>>>(h, eps, agg, n4);
        gine_msg<<<warpBlocksE, blk, 0, stream>>>(h, e16, src, dst, agg, E);
        gemm_bias_act<128, ACT_RELU, false><<<gN, blk, 0, stream>>>(
            agg, w1, DFEAT, b1, t0, nullptr, N);
        if (i < 3)
            gemm_bias_act<128, ACT_RELU, false><<<gN, blk, 0, stream>>>(
                t0, w2, DFEAT, b2, h, nullptr, N);
        else
            gemm_bias_act<128, ACT_NONE, false><<<gN, blk, 0, stream>>>(
                t0, w2, DFEAT, b2, h, nullptr, N);
    }

    // --- final projection, pooling, head ---
    gemm_bias_act<128, ACT_NONE, false><<<gN, blk, 0, stream>>>(
        h, lin_w, DFEAT, lin_b, t0, nullptr, N);

    int pn = NGRAPH * DFEAT;
    zero_f32<<<(pn + 255) / 256, blk, 0, stream>>>(pool, pn);
    pool_scatter<<<warpBlocksN, blk, 0, stream>>>(t0, batch, pool, N);

    gemm_bias_act<128, ACT_RELU, false><<<dim3(NGRAPH / BM, HEADH / BN), blk, 0,
                                          stream>>>(pool, head_w0, HEADH,
                                                    head_b0, hmid, nullptr,
                                                    NGRAPH);
    head_out<<<(NGRAPH * NOUT + 255) / 256, blk, 0, stream>>>(
        hmid, head_w1, head_b1, (float*)d_out);
}
